// QNetwork_41137196761218
// MI455X (gfx1250) — compile-verified, matrix operand order checked by reading, not run on hardware
//
#include <hip/hip_runtime.h>
#include <hip/hip_bf16.h>

// ---------------------------------------------------------------------------
// QNetwork (2x GraphSAGE + MLP head) for MI455X / gfx1250, wave32 + WMMA bf16.
//   N=100000 nodes, E=1.6M edges, D=32, H=256, A=4
// Aggregation is the memory roofline (bf16 gather + f32 atomic scatter into an
// L2-resident accumulator); dense GEMMs run on v_wmma_f32_16x16x32_bf16 with
// fp32 accumulation. Weights are repacked once per launch into fragment-major
// bf16 layout so every fragment (A and B) loads as global_load_b128 pairs.
// ---------------------------------------------------------------------------

typedef __attribute__((ext_vector_type(16))) __bf16 v16bf;
typedef __attribute__((ext_vector_type(8)))  float  v8f;

union FragU { unsigned u[8]; v16bf v; };

__device__ __forceinline__ unsigned f2bf(float f) {
  unsigned u = __builtin_bit_cast(unsigned, f);
  return ((u + 0x7fffu + ((u >> 16) & 1u)) >> 16) & 0xffffu;  // RNE
}
__device__ __forceinline__ unsigned pack2(float lo, float hi) {
  return f2bf(lo) | (f2bf(hi) << 16);
}
__device__ __forceinline__ float bflo(unsigned u) {
  return __builtin_bit_cast(float, u << 16);
}
__device__ __forceinline__ float bfhi(unsigned u) {
  return __builtin_bit_cast(float, u & 0xffff0000u);
}

// ---------------------------------------------------------------- utilities
__global__ void zero4_kernel(float4* __restrict__ p, size_t n4) {
  size_t i = (size_t)blockIdx.x * blockDim.x + threadIdx.x;
  if (i < n4) p[i] = make_float4(0.f, 0.f, 0.f, 0.f);
}

__global__ void degree_kernel(const int* __restrict__ tgt, float* __restrict__ deg, int nE) {
  int e = blockIdx.x * blockDim.x + threadIdx.x;
  if (e < nE) atomicAdd(&deg[tgt[e]], 1.0f);
}

__global__ void invdeg_kernel(float* __restrict__ deg, int n) {
  int i = blockIdx.x * blockDim.x + threadIdx.x;
  if (i < n) {
    float d = deg[i];
    deg[i] = d > 0.f ? 1.f / d : 0.f;   // PyG: zero-degree -> 0
  }
}

// pack fp32 activations [rows,K] -> bf16 pairs [rows, K/2] (optional row scale)
__global__ void pack_rows_bf16(const float* __restrict__ in,
                               const float* __restrict__ rowscale,
                               unsigned* __restrict__ out, int rows, int kp) {
  size_t t = (size_t)blockIdx.x * blockDim.x + threadIdx.x;
  size_t total = (size_t)rows * kp;
  if (t >= total) return;
  int r = (int)(t / kp);
  float s = rowscale ? rowscale[r] : 1.0f;
  float2 v = ((const float2*)in)[t];          // pairs are contiguous in K
  out[t] = pack2(v.x * s, v.y * s);
}

// pack fp32 weights W[Kw][Nw] -> fragment-major bf16 pairs:
//   out[((kb*Nw + c)*2 + h)*8 + i] = {W[2p][c], W[2p+1][c]},
//   p = kb*16 + h*4 + (i<4 ? i : i+4)   (ISA 16-bit B-fragment pair mapping)
// so each lane's 8-word B fragment is contiguous (2 x b128 loads in the GEMM).
__global__ void pack_wfrag_bf16(const float* __restrict__ W, unsigned* __restrict__ out,
                                int Kw, int Nw) {
  size_t t = (size_t)blockIdx.x * blockDim.x + threadIdx.x;
  size_t total = (size_t)(Kw / 32) * Nw * 16;
  if (t >= total) return;
  int kb = (int)(t / ((size_t)Nw * 16));
  int rem = (int)(t % ((size_t)Nw * 16));
  int c = rem >> 4;
  int j = rem & 15;
  int h = j >> 3, i = j & 7;
  int p = kb * 16 + h * 4 + (i < 4 ? i : i + 4);
  out[t] = pack2(W[(size_t)(2 * p) * Nw + c], W[(size_t)(2 * p + 1) * Nw + c]);
}

// ------------------------------------------------------------- aggregation
template <int F>
__global__ void scatter_f32(const float* __restrict__ feat, const int* __restrict__ src,
                            const int* __restrict__ tgt, float* __restrict__ aggr, int nE) {
  constexpr int Q = F / 4;
  size_t t = (size_t)blockIdx.x * blockDim.x + threadIdx.x;
  size_t e = t / Q;
  int q = (int)(t % Q);
  if (e >= (size_t)nE) return;
  int s = src[e], d = tgt[e];
  float4 v = ((const float4*)(feat + (size_t)s * F))[q];
  float* o = aggr + (size_t)d * F + q * 4;
  atomicAdd(o + 0, v.x); atomicAdd(o + 1, v.y);
  atomicAdd(o + 2, v.z); atomicAdd(o + 3, v.w);
}

// gather bf16 pairs (half the read traffic), accumulate fp32 atomics (F = 256)
__global__ void scatter_bf16_256(const unsigned* __restrict__ hb, const int* __restrict__ src,
                                 const int* __restrict__ tgt, float* __restrict__ aggr, int nE) {
  constexpr int Q = 64;  // 256/4 features per thread-chunk
  size_t t = (size_t)blockIdx.x * blockDim.x + threadIdx.x;
  size_t e = t / Q;
  int q = (int)(t % Q);
  if (e >= (size_t)nE) return;
  int s = src[e], d = tgt[e];
  uint2 p = ((const uint2*)(hb + (size_t)s * 128))[q];
  float* o = aggr + (size_t)d * 256 + q * 4;
  atomicAdd(o + 0, bflo(p.x)); atomicAdd(o + 1, bfhi(p.x));
  atomicAdd(o + 2, bflo(p.y)); atomicAdd(o + 3, bfhi(p.y));
}

// -------------------------------------------------------------- WMMA GEMM
// Out = relu( A1@B1 [+ A2@B2] + bias ), fp32 accumulate, bf16 operands.
// A*: packed pairs [nrows, K/2]; B*: fragment-major pack (see pack_wfrag_bf16).
// Block = 256 threads = 8 waves as 2(M) x 4(N) -> 32x64 block tile,
// each wave owns one 16x16 C tile; K stepped by 32 per wmma.
template <int K, int NCOLS, bool DUAL>
__global__ __launch_bounds__(256)
void sage_gemm_kernel(const unsigned* __restrict__ A1, const unsigned* __restrict__ B1,
                      const unsigned* __restrict__ A2, const unsigned* __restrict__ B2,
                      const float* __restrict__ bias, float* __restrict__ Out,
                      int nrows) {
  constexpr int KP = K / 2;                     // pairs per row
  constexpr int KB = K / 32;                    // 32-K blocks
  constexpr int BSTRIDE = NCOLS * 16;           // uints per K-block of B pack
  const int lane = threadIdx.x & 31;
  const int wave = threadIdx.x >> 5;            // 0..7
  const int wm = wave >> 2, wn = wave & 3;      // 2 x 4 wave grid
  const int tileM = blockIdx.x * 32 + wm * 16;
  const int tileN = blockIdx.y * 64 + wn * 16;
  const int l16 = lane & 15;
  const int half = lane >> 4;                   // 0: K-low half, 1: K-high half
  const int kbp = half * 4;                     // pair-index base (ISA A layout)

  int row = tileM + l16;
  int rowc = row < nrows ? row : nrows - 1;     // clamp loads, keep EXEC all-1s
  const int col = tileN + l16;
  const unsigned* a1row = A1 + (size_t)rowc * KP + kbp;
  const unsigned* a2row = DUAL ? (A2 + (size_t)rowc * KP + kbp) : nullptr;
  const unsigned* b1f = B1 + ((size_t)col * 2 + half) * 8;   // fragment-major
  const unsigned* b2f = DUAL ? (B2 + ((size_t)col * 2 + half) * 8) : nullptr;

  v8f c = {};
  for (int kb = 0; kb < KB; ++kb) {             // 32 K-values per iteration
    FragU a, b;
    const unsigned* ap = a1row + kb * 16;       // imm offsets {0..3, 8..11}
    const unsigned* bp = b1f + (size_t)kb * BSTRIDE;  // 8 contiguous words
#pragma unroll
    for (int i = 0; i < 8; ++i) {
      a.u[i] = ap[i < 4 ? i : i + 4];           // -> 2 x global_load_b128
      b.u[i] = bp[i];                           // -> 2 x global_load_b128
    }
    c = __builtin_amdgcn_wmma_f32_16x16x32_bf16(false, a.v, false, b.v,
                                                (short)0, c, false, false);
    if (DUAL) {
      const unsigned* ap2 = a2row + kb * 16;
      const unsigned* bp2 = b2f + (size_t)kb * BSTRIDE;
#pragma unroll
      for (int i = 0; i < 8; ++i) {
        a.u[i] = ap2[i < 4 ? i : i + 4];
        b.u[i] = bp2[i];
      }
      c = __builtin_amdgcn_wmma_f32_16x16x32_bf16(false, a.v, false, b.v,
                                                  (short)0, c, false, false);
    }
  }

  // epilogue: C layout -> lane = N-col, vgpr i = M-row (half*8 + i)
  const float bv = bias[col];
  float* op = Out + (size_t)(tileM + half * 8) * NCOLS + col;
  // tileM is wave-uniform: force a scalar branch (no EXEC churn)
  if (__builtin_amdgcn_readfirstlane(tileM) + 16 <= nrows) {
#pragma unroll
    for (int i = 0; i < 8; ++i) {
      float v = c[i] + bv;
      op[(size_t)i * NCOLS] = v > 0.f ? v : 0.f;   // imm-offset store clause
    }
  } else {
#pragma unroll
    for (int i = 0; i < 8; ++i) {
      if (tileM + half * 8 + i < nrows) {
        float v = c[i] + bv;
        op[(size_t)i * NCOLS] = v > 0.f ? v : 0.f;
      }
    }
  }
}

// ------------------------------------------------------------------- head
// out[r, 0..3] = h3[r, :] @ wh2(256x4) + bh2 ; memory bound on h3 (one pass)
__global__ void head_kernel(const float* __restrict__ h, const float* __restrict__ w,
                            const float* __restrict__ b, float* __restrict__ out, int nrows) {
  int r = blockIdx.x * blockDim.x + threadIdx.x;
  if (r >= nrows) return;
  const float4* hv = (const float4*)(h + (size_t)r * 256);
  const float4* wv = (const float4*)w;          // wv[k] = row k of wh2 (4 actions)
  float a0 = 0.f, a1 = 0.f, a2 = 0.f, a3 = 0.f;
#pragma unroll 4
  for (int k4 = 0; k4 < 64; ++k4) {
    float4 x = hv[k4];
    float4 w0 = wv[k4 * 4 + 0], w1 = wv[k4 * 4 + 1];
    float4 w2 = wv[k4 * 4 + 2], w3 = wv[k4 * 4 + 3];
    a0 += x.x * w0.x + x.y * w1.x + x.z * w2.x + x.w * w3.x;
    a1 += x.x * w0.y + x.y * w1.y + x.z * w2.y + x.w * w3.y;
    a2 += x.x * w0.z + x.y * w1.z + x.z * w2.z + x.w * w3.z;
    a3 += x.x * w0.w + x.y * w1.w + x.z * w2.w + x.w * w3.w;
  }
  float4 bb = *(const float4*)b;
  ((float4*)out)[r] = make_float4(a0 + bb.x, a1 + bb.y, a2 + bb.z, a3 + bb.w);
}

// ---------------------------------------------------------------- launcher
extern "C" void kernel_launch(void* const* d_in, const int* in_sizes, int n_in,
                              void* d_out, int out_size, void* d_ws, size_t ws_size,
                              hipStream_t stream) {
  const float* x   = (const float*)d_in[0];
  const int*   ei  = (const int*)d_in[1];       // [2,E] int32
  const float* w1l = (const float*)d_in[2];
  const float* b1l = (const float*)d_in[3];
  const float* w1r = (const float*)d_in[4];
  const float* w2l = (const float*)d_in[5];
  const float* b2l = (const float*)d_in[6];
  const float* w2r = (const float*)d_in[7];
  const float* wh1 = (const float*)d_in[8];
  const float* bh1 = (const float*)d_in[9];
  const float* wh2 = (const float*)d_in[10];
  const float* bh2 = (const float*)d_in[11];
  float* out = (float*)d_out;

  const int n  = in_sizes[0] / 32;              // D = 32
  const int nE = in_sizes[1] / 2;
  const int* src = ei;
  const int* tgt = ei + nE;

  // ---- scratch layout (~385 MB total), 256B aligned regions -------------
  char* wsp = (char*)d_ws;
  auto alloc = [&](size_t bytes) -> char* {
    char* p = wsp;
    wsp += (bytes + 255) & ~(size_t)255;
    return p;
  };
  float*    deg   = (float*)alloc((size_t)n * 4);            // -> inv_deg in place
  float*    aggr1 = (float*)alloc((size_t)n * 32 * 4);
  float*    aggr2 = (float*)alloc((size_t)n * 256 * 4);
  float*    h1f   = (float*)alloc((size_t)n * 256 * 4);
  unsigned* h1b   = (unsigned*)alloc((size_t)n * 128 * 4);
  unsigned* a2b   = (unsigned*)alloc((size_t)n * 128 * 4);
  unsigned* h2b   = (unsigned*)alloc((size_t)n * 128 * 4);
  unsigned* xb    = (unsigned*)alloc((size_t)n * 16 * 4);
  unsigned* a1b   = (unsigned*)alloc((size_t)n * 16 * 4);
  unsigned* w1lp  = (unsigned*)alloc((size_t)16 * 256 * 4);
  unsigned* w1rp  = (unsigned*)alloc((size_t)16 * 256 * 4);
  unsigned* w2lp  = (unsigned*)alloc((size_t)128 * 256 * 4);
  unsigned* w2rp  = (unsigned*)alloc((size_t)128 * 256 * 4);
  unsigned* wh1p  = (unsigned*)alloc((size_t)128 * 256 * 4);
  float* h2f = h1f;    // h1f dead after pack -> reuse for h2
  float* h3f = aggr2;  // aggr2 dead after pack -> reuse for h3

  auto cdiv = [](size_t a, size_t b) { return (unsigned)((a + b - 1) / b); };

  // zero [deg .. aggr2] (contiguous, all multiples of 16B)
  size_t zbytes = (size_t)((char*)aggr2 - (char*)deg) + (size_t)n * 256 * 4;
  size_t z4 = zbytes / 16;
  zero4_kernel<<<cdiv(z4, 256), 256, 0, stream>>>((float4*)deg, z4);

  // degree -> inv_deg
  degree_kernel<<<cdiv(nE, 256), 256, 0, stream>>>(tgt, deg, nE);
  invdeg_kernel<<<cdiv(n, 256), 256, 0, stream>>>(deg, n);

  // weight packs into fragment-major layout (tiny, L2 resident afterwards)
  pack_wfrag_bf16<<<cdiv((size_t)1 * 256 * 16, 256), 256, 0, stream>>>(w1l, w1lp, 32, 256);
  pack_wfrag_bf16<<<cdiv((size_t)1 * 256 * 16, 256), 256, 0, stream>>>(w1r, w1rp, 32, 256);
  pack_wfrag_bf16<<<cdiv((size_t)8 * 256 * 16, 256), 256, 0, stream>>>(w2l, w2lp, 256, 256);
  pack_wfrag_bf16<<<cdiv((size_t)8 * 256 * 16, 256), 256, 0, stream>>>(w2r, w2rp, 256, 256);
  pack_wfrag_bf16<<<cdiv((size_t)8 * 256 * 16, 256), 256, 0, stream>>>(wh1, wh1p, 256, 256);

  // ---- layer 1 ----------------------------------------------------------
  pack_rows_bf16<<<cdiv((size_t)n * 16, 256), 256, 0, stream>>>(x, nullptr, xb, n, 16);
  scatter_f32<32><<<cdiv((size_t)nE * 8, 256), 256, 0, stream>>>(x, src, tgt, aggr1, nE);
  pack_rows_bf16<<<cdiv((size_t)n * 16, 256), 256, 0, stream>>>(aggr1, deg, a1b, n, 16);
  {
    dim3 g(cdiv(n, 32), 4);
    sage_gemm_kernel<32, 256, true><<<g, 256, 0, stream>>>(a1b, w1lp, xb, w1rp, b1l, h1f, n);
  }
  pack_rows_bf16<<<cdiv((size_t)n * 128, 256), 256, 0, stream>>>(h1f, nullptr, h1b, n, 128);

  // ---- layer 2 ----------------------------------------------------------
  scatter_bf16_256<<<cdiv((size_t)nE * 64, 256), 256, 0, stream>>>(h1b, src, tgt, aggr2, nE);
  pack_rows_bf16<<<cdiv((size_t)n * 128, 256), 256, 0, stream>>>(aggr2, deg, a2b, n, 128);
  {
    dim3 g(cdiv(n, 32), 4);
    sage_gemm_kernel<256, 256, true><<<g, 256, 0, stream>>>(a2b, w2lp, h1b, w2rp, b2l, h2f, n);
  }
  pack_rows_bf16<<<cdiv((size_t)n * 128, 256), 256, 0, stream>>>(h2f, nullptr, h2b, n, 128);

  // ---- MLP head ---------------------------------------------------------
  {
    dim3 g(cdiv(n, 32), 4);
    sage_gemm_kernel<256, 256, false><<<g, 256, 0, stream>>>(h2b, wh1p, nullptr, nullptr, bh1, h3f, n);
  }
  head_kernel<<<cdiv(n, 256), 256, 0, stream>>>(h3f, wh2, bh2, out, n);
}
